// ind_lstm_73108933313080
// MI455X (gfx1250) — matrix operand order repaired and checked by reading, not compiled, not run on hardware
//
#include <hip/hip_runtime.h>
#include <hip/hip_bf16.h>

#define T_STEPS 4096
#define BLOCK_N 1024
#define SIG_N   1024
#define G4      4096   // 4*SIG = 4*BLOCK
#define LAM     0.25f

typedef __bf16 bf16x16 __attribute__((ext_vector_type(16)));
typedef __bf16 bf16x8  __attribute__((ext_vector_type(8)));
typedef float  f32x8   __attribute__((ext_vector_type(8)));

__device__ __forceinline__ __bf16 f2bf(float f) {
  unsigned u = __builtin_bit_cast(unsigned, f);
  u += 0x7FFFu + ((u >> 16) & 1u);            // round-to-nearest-even
  unsigned short h = (unsigned short)(u >> 16);
  return __builtin_bit_cast(__bf16, h);
}

__device__ __forceinline__ float sigm(float x) {
  return 1.0f / (1.0f + __expf(-x));
}

// ---------------------------------------------------------------- init ----
__global__ void init_state(float* hbuf, float* gatebuf, float* l1p, float* msep,
                           unsigned* bar) {
  int i = threadIdx.x;
  for (int k = i; k < SIG_N; k += blockDim.x) hbuf[k] = 0.0f;
  for (int k = i; k < G4;    k += blockDim.x) gatebuf[k] = 0.0f;
  if (i < 32) l1p[i] = 0.0f;
  if (i < 8)  msep[i] = 0.0f;
  if (i < 2)  bar[i] = 0u;
}

// --------------------------------------------------------- bias fusion ----
__global__ void prep_bias(const float* __restrict__ e0, const float* __restrict__ e1,
                          const float* __restrict__ d0, const float* __restrict__ d1,
                          float* __restrict__ biasE, float* __restrict__ biasD,
                          float* __restrict__ biasZ) {
  int i = blockIdx.x * blockDim.x + threadIdx.x;
  if (i < G4) {
    biasE[i] = e0[i] + e1[i];
    biasD[i] = d0[i] + d1[i];
    biasZ[i] = 0.0f;
  }
}

// ------------------------------------------------------------- convert ----
__global__ void cvt_f32_to_bf16(const float* __restrict__ src,
                                __bf16* __restrict__ dst, int n) {
  int i = blockIdx.x * blockDim.x + threadIdx.x;
  int stride = gridDim.x * blockDim.x;
  for (; i < n; i += stride) dst[i] = f2bf(src[i]);
}

// ------------------------------------------------------- WMMA bf16 GEMM ---
// C[M,N] = A[M,K](bf16, row-major) * B[N,K](bf16, row-major)^T + bias[n]
// Block: 256 threads = 8 waves; wave tile 32x64; block tile 128x128.
__global__ __launch_bounds__(256)
void wmma_gemm_nt(const __bf16* __restrict__ A, const __bf16* __restrict__ B,
                  const float* __restrict__ bias,
                  float* __restrict__ C, int M, int N, int K) {
  const int lane  = threadIdx.x & 31;
  const int wave  = threadIdx.x >> 5;   // 0..7
  const int wr    = wave >> 1;          // 0..3 (M)
  const int wc    = wave & 1;           // 0..1 (N)
  const int mBase = blockIdx.y * 128 + wr * 32;
  const int nBase = blockIdx.x * 128 + wc * 64;
  const int lhalf = lane >> 4;          // 0 or 1
  const int l16   = lane & 15;

  f32x8 acc[2][4];
#pragma unroll
  for (int mi = 0; mi < 2; ++mi)
#pragma unroll
    for (int ni = 0; ni < 4; ++ni)
      acc[mi][ni] = (f32x8){0.f,0.f,0.f,0.f,0.f,0.f,0.f,0.f};

  for (int kt = 0; kt < K; kt += 32) {
    bf16x16 afrag[2], bfrag[4];
    // A fragment: lanes 0-15 -> K kt..kt+15,
    //             lanes 16-31 -> K kt+16..kt+31
#pragma unroll
    for (int mi = 0; mi < 2; ++mi) {
      const __bf16* ap = A + (size_t)(mBase + mi * 16 + l16) * K + kt + lhalf * 16;
      bf16x8 lo = *(const bf16x8*)(ap);
      bf16x8 hi = *(const bf16x8*)(ap + 8);
      afrag[mi] = __builtin_shufflevector(lo, hi, 0,1,2,3,4,5,6,7,8,9,10,11,12,13,14,15);
      __builtin_prefetch(ap + 32, 0, 3);           // next k-tile
    }
    // B fragment: lane n = l16, lanes 0-15 hold K kt..kt+15, lanes 16-31 hold kt+16..kt+31
#pragma unroll
    for (int ni = 0; ni < 4; ++ni) {
      const __bf16* bp = B + (size_t)(nBase + ni * 16 + l16) * K + kt + lhalf * 16;
      bfrag[ni] = *(const bf16x16*)(bp);
      __builtin_prefetch(bp + 32, 0, 3);           // next k-tile
    }
#pragma unroll
    for (int mi = 0; mi < 2; ++mi)
#pragma unroll
      for (int ni = 0; ni < 4; ++ni)
        acc[mi][ni] = __builtin_amdgcn_wmma_f32_16x16x32_bf16(
            false, afrag[mi], false, bfrag[ni], (short)0, acc[mi][ni], false, false);
  }

  // Epilogue: C/D layout — VGPR r, lanes 0-15: (M=r, N=lane); lanes 16-31: (M=r+8, N=lane-16)
#pragma unroll
  for (int mi = 0; mi < 2; ++mi) {
#pragma unroll
    for (int ni = 0; ni < 4; ++ni) {
      const int col = nBase + ni * 16 + l16;
      const float b = bias[col];                   // always valid, no branch
#pragma unroll
      for (int r = 0; r < 8; ++r) {
        const int row = mBase + mi * 16 + r + 8 * lhalf;
        C[(size_t)row * N + col] = acc[mi][ni][r] + b;
      }
    }
  }
}

// ------------------------------------------------------ grid barrier ------
__device__ __forceinline__ void grid_barrier(unsigned* cnt, unsigned* gen, unsigned nwg) {
  __syncthreads();
  if (threadIdx.x == 0) {
    __threadfence();
    unsigned g = __hip_atomic_load(gen, __ATOMIC_ACQUIRE, __HIP_MEMORY_SCOPE_AGENT);
    unsigned arrived = __hip_atomic_fetch_add(cnt, 1u, __ATOMIC_ACQ_REL, __HIP_MEMORY_SCOPE_AGENT);
    if (arrived == nwg - 1u) {
      __hip_atomic_store(cnt, 0u, __ATOMIC_RELAXED, __HIP_MEMORY_SCOPE_AGENT);
      __hip_atomic_fetch_add(gen, 1u, __ATOMIC_ACQ_REL, __HIP_MEMORY_SCOPE_AGENT);
    } else {
      while (__hip_atomic_load(gen, __ATOMIC_ACQUIRE, __HIP_MEMORY_SCOPE_AGENT) == g) {
        __builtin_amdgcn_s_sleep(1);
      }
    }
  }
  __syncthreads();
}

// --------------------------------------------------- encoder serial scan --
// 32 blocks x 128 threads = 4096 threads; thread gid owns gate row `gid`.
// Whh (fp32, 16 MB) streams from L2 each step; h (4 KB) staged into LDS per block.
__global__ __launch_bounds__(128)
void enc_scan(const float* __restrict__ Genc,     // [T][4096] x-part gates + biases
              const float* __restrict__ Whh,      // [4096][1024] fp32
              float* __restrict__ hbuf,           // [1024]
              float* __restrict__ gatebuf,        // [4096]
              __bf16* __restrict__ sigbf,         // [T][1024] signal in bf16
              float* __restrict__ l1_partial,     // [32]
              unsigned* __restrict__ bar,         // cnt, gen
              int nsteps, unsigned nwg) {
  const int gid = blockIdx.x * blockDim.x + threadIdx.x;     // 0..4095
  const float4* __restrict__ w4 = (const float4*)(Whh + (size_t)gid * SIG_N);
  float c  = 0.0f;   // valid when gid < 1024
  float l1 = 0.0f;

  __shared__ float hsh[SIG_N];                      // 4 KB of the 320 KB WGP LDS
  __shared__ float red[128];

  for (int t = 0; t < nsteps; ++t) {
    // stage h (valid after previous step's release barrier) into LDS
    {
      const float4* __restrict__ hg = (const float4*)hbuf;
      float4* hs = (float4*)hsh;
      for (int k = threadIdx.x; k < SIG_N / 4; k += blockDim.x) hs[k] = hg[k];
    }
    __syncthreads();

    // recurrent matvec: one gate per thread, h from LDS
    float s = Genc[(size_t)t * G4 + gid];
    const float4* h4 = (const float4*)hsh;
#pragma unroll 4
    for (int k = 0; k < SIG_N / 4; ++k) {
      float4 w = w4[k], h = h4[k];
      s += w.x * h.x + w.y * h.y + w.z * h.z + w.w * h.w;
    }
    gatebuf[gid] = s;
    grid_barrier(bar, bar + 1, nwg);               // gates visible everywhere

    if (gid < SIG_N) {
      float iv = sigm(gatebuf[gid]);
      float fv = sigm(gatebuf[SIG_N + gid]);
      float gv = tanhf(gatebuf[2 * SIG_N + gid]);
      float ov = sigm(gatebuf[3 * SIG_N + gid]);
      c = fv * c + iv * gv;
      float h = ov * tanhf(c);
      hbuf[gid] = h;
      sigbf[(size_t)t * SIG_N + gid] = f2bf(h);
      l1 += fabsf(h);
    }
    grid_barrier(bar, bar + 1, nwg);               // h visible before next step
  }

  red[threadIdx.x] = l1;
  __syncthreads();
  for (int s = 64; s > 0; s >>= 1) {
    if (threadIdx.x < s) red[threadIdx.x] += red[threadIdx.x + s];
    __syncthreads();
  }
  if (threadIdx.x == 0) l1_partial[blockIdx.x] = red[0];
}

// ------------------------------------------------- decoder parallel scan --
// Per-unit independent: 8 blocks x 128 threads, thread j owns column j, c in a register.
__global__ __launch_bounds__(128)
void dec_scan(const float* __restrict__ Gx,       // [T][4096], rows 0..T-2 used
              const float* __restrict__ Gh,       // [T][4096], rows 1..T-1 used
              const float* __restrict__ x,        // inputs [T][1024]
              float* __restrict__ mse_partial,    // [8]
              int nsteps) {
  const int j = blockIdx.x * blockDim.x + threadIdx.x;       // 0..1023
  float c = 0.0f, acc = 0.0f;
  for (int t = 1; t < nsteps; ++t) {
    const float* gx = Gx + (size_t)(t - 1) * G4;
    const float* gh = Gh + (size_t)t * G4;
    float iv = sigm(gx[j]                 + gh[j]);
    float fv = sigm(gx[BLOCK_N + j]       + gh[BLOCK_N + j]);
    float gv = tanhf(gx[2 * BLOCK_N + j]  + gh[2 * BLOCK_N + j]);
    float ov = sigm(gx[3 * BLOCK_N + j]   + gh[3 * BLOCK_N + j]);
    c = fv * c + iv * gv;
    float h = ov * tanhf(c);
    float d = h - x[(size_t)t * BLOCK_N + j];
    acc += d * d;
  }
  __shared__ float red[128];
  red[threadIdx.x] = acc;
  __syncthreads();
  for (int s = 64; s > 0; s >>= 1) {
    if (threadIdx.x < s) red[threadIdx.x] += red[threadIdx.x + s];
    __syncthreads();
  }
  if (threadIdx.x == 0) mse_partial[blockIdx.x] = red[0];
}

// ---------------------------------------------------------------- final ---
__global__ void finalize(const float* __restrict__ l1p, const float* __restrict__ msep,
                         float* __restrict__ out) {
  if (threadIdx.x == 0 && blockIdx.x == 0) {
    float l1 = 0.0f;
    for (int i = 0; i < 32; ++i) l1 += l1p[i];          // fixed order: deterministic
    float mse = 0.0f;
    for (int i = 0; i < 8; ++i) mse += msep[i];
    mse /= (float)((T_STEPS - 1) * BLOCK_N);
    l1  /= (float)(T_STEPS * SIG_N);
    out[0] = mse + LAM * l1;
    out[1] = mse;
    out[2] = l1;
  }
}

// ---------------------------------------------------------------------------
extern "C" void kernel_launch(void* const* d_in, const int* in_sizes, int n_in,
                              void* d_out, int out_size, void* d_ws, size_t ws_size,
                              hipStream_t stream) {
  const float* inputs  = (const float*)d_in[0];   // [T][1024]
  const float* eWih    = (const float*)d_in[1];   // [4096][1024]
  const float* eWhh    = (const float*)d_in[2];   // [4096][1024]
  const float* ebih    = (const float*)d_in[3];   // [4096]
  const float* ebhh    = (const float*)d_in[4];   // [4096]
  const float* dWih    = (const float*)d_in[5];   // [4096][1024]
  const float* dWhh    = (const float*)d_in[6];   // [4096][1024]
  const float* dbih    = (const float*)d_in[7];   // [4096]
  const float* dbhh    = (const float*)d_in[8];   // [4096]
  float* out = (float*)d_out;

  // workspace carve-up (256-byte aligned)
  char* ws = (char*)d_ws;
  size_t off = 0;
  auto carve = [&](size_t bytes) -> char* {
    char* p = ws + off;
    off = (off + bytes + 255) & ~(size_t)255;
    return p;
  };
  const size_t NBF = (size_t)T_STEPS * BLOCK_N;        // 4M elements
  __bf16* Xbf    = (__bf16*)carve(NBF * 2);
  __bf16* eWihbf = (__bf16*)carve(NBF * 2);
  __bf16* dWihbf = (__bf16*)carve(NBF * 2);
  __bf16* dWhhbf = (__bf16*)carve(NBF * 2);
  __bf16* sigbf  = (__bf16*)carve(NBF * 2);
  float*  Genc   = (float*)carve((size_t)T_STEPS * G4 * 4);   // reused for Gh_dec
  float*  Gxdec  = (float*)carve((size_t)T_STEPS * G4 * 4);
  float*  biasE  = (float*)carve(G4 * 4);
  float*  biasD  = (float*)carve(G4 * 4);
  float*  biasZ  = (float*)carve(G4 * 4);
  float*  hbuf   = (float*)carve(SIG_N * 4);
  float*  gateb  = (float*)carve(G4 * 4);
  float*  l1p    = (float*)carve(32 * 4);
  float*  msep   = (float*)carve(8 * 4);
  unsigned* bar  = (unsigned*)carve(8);
  (void)ws_size; (void)in_sizes; (void)n_in; (void)out_size;

  init_state<<<1, 256, 0, stream>>>(hbuf, gateb, l1p, msep, bar);
  prep_bias<<<G4 / 256, 256, 0, stream>>>(ebih, ebhh, dbih, dbhh, biasE, biasD, biasZ);

  const int nconv = (int)NBF;
  cvt_f32_to_bf16<<<2048, 256, 0, stream>>>(inputs, Xbf,    nconv);
  cvt_f32_to_bf16<<<2048, 256, 0, stream>>>(eWih,   eWihbf, nconv);
  cvt_f32_to_bf16<<<2048, 256, 0, stream>>>(dWih,   dWihbf, nconv);
  cvt_f32_to_bf16<<<2048, 256, 0, stream>>>(dWhh,   dWhhbf, nconv);

  dim3 ggrid(G4 / 128, T_STEPS / 128);
  // encoder x-part gates: Genc = X * eWih^T + (ebih+ebhh)
  wmma_gemm_nt<<<ggrid, 256, 0, stream>>>(Xbf, eWihbf, biasE, Genc,
                                          T_STEPS, G4, BLOCK_N);
  // decoder x-part gates: Gxdec = X * dWih^T + (dbih+dbhh)
  wmma_gemm_nt<<<ggrid, 256, 0, stream>>>(Xbf, dWihbf, biasD, Gxdec,
                                          T_STEPS, G4, SIG_N);

  // serial encoder recurrence (32 co-resident workgroups, grid barrier)
  enc_scan<<<32, 128, 0, stream>>>(Genc, eWhh, hbuf, gateb, sigbf, l1p, bar,
                                   T_STEPS, 32u);

  // decoder h-part gates: Gh = signal * dWhh^T  (reuse Genc buffer)
  wmma_gemm_nt<<<ggrid, 256, 0, stream>>>(sigbf, dWhhbf, biasZ, Genc,
                                          T_STEPS, G4, BLOCK_N);

  // decoder scan: per-unit independent, c in registers
  dec_scan<<<8, 128, 0, stream>>>(Gxdec, Genc, inputs, msep, T_STEPS);

  finalize<<<1, 32, 0, stream>>>(l1p, msep, out);
}
